// HydraConv_25323127177746
// MI455X (gfx1250) — matrix operand order repaired and checked by reading, not compiled, not run on hardware
//
#include <hip/hip_runtime.h>
#include <hip/hip_bf16.h>
#include <stdint.h>

// ---------------------------------------------------------------------------
// HydraConv on MI455X (gfx1250): LayerNorm -> GEMM(bf16 WMMA) -> cosine attn
// pooling -> folded GEMM(bf16 WMMA) + bias + residual.
// GEMM: 256-thread blocks (8 waves), block tile 128x128, wave tile 32x64
// (2x4 v_wmma_f32_16x16x32_bf16 per K-step), direct L2-resident fragment
// loads (global_load_b128) + global_prefetch of the next K-slice.
// ---------------------------------------------------------------------------

typedef __bf16 bf16_t;
typedef __attribute__((ext_vector_type(16))) __bf16    bf16x16;
typedef __attribute__((ext_vector_type(8)))  float     f32x8;
typedef __attribute__((ext_vector_type(4)))  float     f32x4;
typedef __attribute__((ext_vector_type(4)))  uint32_t  u32x4;

#define NB   16          // batch
#define NC   384         // channels C (= GEMM K)
#define NN   4096        // spatial H*W
#define OC3  1152        // 3*C
#define CN   (NC * NN)   // 1,572,864 elements per sample

// ---------------------------------------------------------------------------
// WMMA fragment helpers (ISA 7.12.2 layouts, wave32)
// ---------------------------------------------------------------------------
union FragAB { u32x4 q[2]; bf16x16 v; };

static __device__ __forceinline__ f32x8 wmma_bf16(const FragAB& a, const FragAB& b, f32x8 c) {
    // v_wmma_f32_16x16x32_bf16  D = A(16x32) * B(32x16) + C
    return __builtin_amdgcn_wmma_f32_16x16x32_bf16(false, a.v, false, b.v,
                                                   (short)0, c, false, false);
}

struct Acc { f32x8 c[2][4]; };

// A: row-major (M x 384) bf16, rows = output channels.
// B: row-major (N x 384) bf16, row n = activation column n (K contiguous).
// Each wave computes a 32(M) x 64(N) tile at (m_base, n_base).
static __device__ __forceinline__ void gemm_tile_core(const bf16_t* __restrict__ Abase,
                                                      const bf16_t* __restrict__ Bbase,
                                                      int m_base, int n_base, Acc& acc) {
    const int lane = threadIdx.x & 31;
    const int lr   = lane & 15;
    const int kbA  = (lane < 16) ? 0 : 16;   // A: K-halves {0..15} / {16..31}
    const int kbB  = (lane < 16) ? 0 : 16;   // B: K-halves {0..15} / {16..31}

    const bf16_t* ap[2];
    const bf16_t* bp[4];
#pragma unroll
    for (int mt = 0; mt < 2; ++mt)
        ap[mt] = Abase + (size_t)(m_base + mt * 16 + lr) * NC + kbA;
#pragma unroll
    for (int nt = 0; nt < 4; ++nt)
        bp[nt] = Bbase + (size_t)(n_base + nt * 16 + lr) * NC + kbB;

#pragma unroll 2
    for (int kk = 0; kk < NC; kk += 32) {
        FragAB a[2], b[4];
#pragma unroll
        for (int mt = 0; mt < 2; ++mt) {
            const u32x4* p = (const u32x4*)(ap[mt] + kk);
            a[mt].q[0] = p[0];
            a[mt].q[1] = p[1];                 // 16 contiguous bf16
        }
#pragma unroll
        for (int nt = 0; nt < 4; ++nt) {
            const u32x4* p = (const u32x4*)(bp[nt] + kk);
            b[nt].q[0] = p[0];
            b[nt].q[1] = p[1];                 // 16 contiguous bf16
        }

        if (kk + 32 < NC) {                    // global_prefetch_b8 next K-slice
            __builtin_prefetch((const void*)(ap[0] + kk + 32), 0, 1);
            __builtin_prefetch((const void*)(bp[0] + kk + 32), 0, 1);
        }

#pragma unroll
        for (int mt = 0; mt < 2; ++mt)
#pragma unroll
            for (int nt = 0; nt < 4; ++nt)
                acc.c[mt][nt] = wmma_bf16(a[mt], b[nt], acc.c[mt][nt]);
    }
}

// ---------------------------------------------------------------------------
// 1) per-sample mean / rstd over C*H*W
// ---------------------------------------------------------------------------
__global__ __launch_bounds__(256) void k_stats(const float* __restrict__ X,
                                               float* __restrict__ stats) {
    const int b   = blockIdx.x;
    const int tid = threadIdx.x;
    const f32x4* p = (const f32x4*)(X + (size_t)b * CN);
    float s = 0.f, s2 = 0.f;
    for (int i = tid; i < CN / 4; i += 256) {
        f32x4 v = p[i];
        s  += v[0] + v[1] + v[2] + v[3];
        s2 += v[0]*v[0] + v[1]*v[1] + v[2]*v[2] + v[3]*v[3];
    }
    __shared__ float r0[256], r1[256];
    r0[tid] = s; r1[tid] = s2;
    __syncthreads();
    for (int st = 128; st > 0; st >>= 1) {
        if (tid < st) { r0[tid] += r0[tid + st]; r1[tid] += r1[tid + st]; }
        __syncthreads();
    }
    if (tid == 0) {
        const float inv = 1.0f / (float)CN;
        float mu  = r0[0] * inv;
        float var = r1[0] * inv - mu * mu;
        stats[b]      = mu;
        stats[16 + b] = rsqrtf(var + 1e-5f);
    }
}

// ---------------------------------------------------------------------------
// 2) normalize + affine, transpose (c,n)->(n,c), cast to bf16
// ---------------------------------------------------------------------------
__global__ __launch_bounds__(256) void k_norm_t(const float* __restrict__ X,
                                                const float* __restrict__ gamma,
                                                const float* __restrict__ beta,
                                                const float* __restrict__ stats,
                                                bf16_t* __restrict__ xn) {
    __shared__ float tile[32][33];
    const int b  = blockIdx.z;
    const int c0 = blockIdx.y * 32;
    const int n0 = blockIdx.x * 32;
    const int tx = threadIdx.x & 31;
    const int ty = threadIdx.x >> 5;   // 0..7
    const float mu   = stats[b];
    const float rstd = stats[16 + b];
    const float* Xb  = X + (size_t)b * CN;
#pragma unroll
    for (int i = 0; i < 4; ++i) {
        int c = c0 + ty + i * 8;
        float x = Xb[(size_t)c * NN + n0 + tx];
        tile[ty + i * 8][tx] = (x - mu) * rstd * gamma[c] + beta[c];
    }
    __syncthreads();
    bf16_t* dst = xn + (size_t)b * NN * NC;
#pragma unroll
    for (int i = 0; i < 4; ++i) {
        int n = n0 + ty + i * 8;
        dst[(size_t)n * NC + c0 + tx] = (bf16_t)tile[tx][ty + i * 8];
    }
}

// ---------------------------------------------------------------------------
// 3) cast w_in to bf16
// ---------------------------------------------------------------------------
__global__ __launch_bounds__(256) void k_cast_win(const float* __restrict__ w,
                                                  bf16_t* __restrict__ wb, int n) {
    int i = blockIdx.x * 256 + threadIdx.x;
    if (i < n) wb[i] = (bf16_t)w[i];
}

// ---------------------------------------------------------------------------
// 4) GEMM1: qkv[b,o,n] = w_in[o,:] . Xn[b,n,:]   (bf16 WMMA, fp32 acc)
// ---------------------------------------------------------------------------
__global__ __launch_bounds__(256) void k_gemm1(const bf16_t* __restrict__ win,
                                               const bf16_t* __restrict__ xn,
                                               bf16_t* __restrict__ qkv) {
    const int b    = blockIdx.z;
    const int wave = threadIdx.x >> 5;
    const int m_base = blockIdx.y * 128 + (wave >> 1) * 32;
    const int n_base = blockIdx.x * 128 + (wave & 1)  * 64;
    const bf16_t* Bb = xn + (size_t)b * NN * NC;

    Acc acc = {};
    gemm_tile_core(win, Bb, m_base, n_base, acc);

    bf16_t* Db = qkv + (size_t)b * OC3 * NN;
    const int lane = threadIdx.x & 31;
    const int lr   = lane & 15;
    const int rofs = (lane < 16) ? 0 : 8;
#pragma unroll
    for (int mt = 0; mt < 2; ++mt)
#pragma unroll
        for (int nt = 0; nt < 4; ++nt)
#pragma unroll
            for (int r = 0; r < 8; ++r) {
                int row = m_base + mt * 16 + rofs + r;
                int col = n_base + nt * 16 + lr;
                Db[(size_t)row * NN + col] = (bf16_t)acc.c[mt][nt][r];
            }
}

// ---------------------------------------------------------------------------
// 5) per-(b,n) channel norms of Q and K
// ---------------------------------------------------------------------------
__global__ __launch_bounds__(256) void k_qknorm(const bf16_t* __restrict__ qkv,
                                                float* __restrict__ qnorm,
                                                float* __restrict__ knorm) {
    const int b = blockIdx.y;
    const int n = blockIdx.x * 256 + threadIdx.x;
    const bf16_t* Qb = qkv + (size_t)b * OC3 * NN;
    const bf16_t* Kb = Qb + (size_t)NC * NN;
    float sq = 0.f, sk = 0.f;
    for (int c = 0; c < NC; ++c) {
        float q = (float)Qb[(size_t)c * NN + n];
        float k = (float)Kb[(size_t)c * NN + n];
        sq += q * q;
        sk += k * k;
    }
    qnorm[(size_t)b * NN + n] = rsqrtf(sq + 1e-6f);
    knorm[(size_t)b * NN + n] = rsqrtf(sk + 1e-6f);
}

// ---------------------------------------------------------------------------
// 6) A[b,c] = sum_n Khat[c,n] * V[c,n]
// ---------------------------------------------------------------------------
__global__ __launch_bounds__(256) void k_avec(const bf16_t* __restrict__ qkv,
                                              const float* __restrict__ knorm,
                                              float* __restrict__ Avec) {
    const int c = blockIdx.x;
    const int b = blockIdx.y;
    const bf16_t* Kb = qkv + (size_t)b * OC3 * NN + (size_t)(NC + c) * NN;
    const bf16_t* Vb = qkv + (size_t)b * OC3 * NN + (size_t)(2 * NC + c) * NN;
    const float* kn = knorm + (size_t)b * NN;
    float s = 0.f;
    for (int n = threadIdx.x; n < NN; n += 256)
        s += (float)Kb[n] * kn[n] * (float)Vb[n];
    __shared__ float red[256];
    red[threadIdx.x] = s;
    __syncthreads();
    for (int st = 128; st > 0; st >>= 1) {
        if (threadIdx.x < st) red[threadIdx.x] += red[threadIdx.x + st];
        __syncthreads();
    }
    if (threadIdx.x == 0) Avec[(size_t)b * NC + c] = red[0];
}

// ---------------------------------------------------------------------------
// 7) B2[b,n,c] = Qhat (transpose + qnorm scale, bf16) -- reuses xn buffer
// ---------------------------------------------------------------------------
__global__ __launch_bounds__(256) void k_b2t(const bf16_t* __restrict__ qkv,
                                             const float* __restrict__ qnorm,
                                             bf16_t* __restrict__ b2) {
    __shared__ float tile[32][33];
    const int b  = blockIdx.z;
    const int c0 = blockIdx.y * 32;
    const int n0 = blockIdx.x * 32;
    const int tx = threadIdx.x & 31;
    const int ty = threadIdx.x >> 5;
    const bf16_t* Qb = qkv + (size_t)b * OC3 * NN;
    const float* qn  = qnorm + (size_t)b * NN;
#pragma unroll
    for (int i = 0; i < 4; ++i) {
        int c = c0 + ty + i * 8;
        tile[ty + i * 8][tx] = (float)Qb[(size_t)c * NN + n0 + tx] * qn[n0 + tx];
    }
    __syncthreads();
    bf16_t* dst = b2 + (size_t)b * NN * NC;
#pragma unroll
    for (int i = 0; i < 4; ++i) {
        int n = n0 + ty + i * 8;
        dst[(size_t)n * NC + c0 + tx] = (bf16_t)tile[tx][ty + i * 8];
    }
}

// ---------------------------------------------------------------------------
// 8) w2[b,o,c] = w_out[o,c] * A[b,c]   (fold the pooled vector into weights)
// ---------------------------------------------------------------------------
__global__ __launch_bounds__(256) void k_w2(const float* __restrict__ w_out,
                                            const float* __restrict__ Avec,
                                            bf16_t* __restrict__ w2) {
    int i = blockIdx.x * 256 + threadIdx.x;
    if (i >= NB * NC * NC) return;
    int b = i / (NC * NC);
    int rem = i - b * (NC * NC);
    int c = rem % NC;
    w2[i] = (bf16_t)(w_out[rem] * Avec[(size_t)b * NC + c]);
}

// ---------------------------------------------------------------------------
// 9) GEMM2: out[b,o,n] = w2[b,o,:] . Qhat[b,n,:] + b_out[o] + X[b,o,n]
// ---------------------------------------------------------------------------
__global__ __launch_bounds__(256) void k_gemm2(const bf16_t* __restrict__ w2,
                                               const bf16_t* __restrict__ b2,
                                               const float* __restrict__ b_out,
                                               const float* __restrict__ X,
                                               float* __restrict__ out) {
    const int b    = blockIdx.z;
    const int wave = threadIdx.x >> 5;
    const int m_base = blockIdx.y * 128 + (wave >> 1) * 32;
    const int n_base = blockIdx.x * 128 + (wave & 1)  * 64;
    const bf16_t* Ab = w2 + (size_t)b * NC * NC;
    const bf16_t* Bb = b2 + (size_t)b * NN * NC;

    Acc acc = {};
    gemm_tile_core(Ab, Bb, m_base, n_base, acc);

    const float* Xb = X + (size_t)b * CN;
    float* Ob = out + (size_t)b * CN;
    const int lane = threadIdx.x & 31;
    const int lr   = lane & 15;
    const int rofs = (lane < 16) ? 0 : 8;
#pragma unroll
    for (int mt = 0; mt < 2; ++mt)
#pragma unroll
        for (int nt = 0; nt < 4; ++nt)
#pragma unroll
            for (int r = 0; r < 8; ++r) {
                int row = m_base + mt * 16 + rofs + r;
                int col = n_base + nt * 16 + lr;
                size_t idx = (size_t)row * NN + col;
                Ob[idx] = acc.c[mt][nt][r] + b_out[row] + Xb[idx];
            }
}

// ---------------------------------------------------------------------------
// launch
// ---------------------------------------------------------------------------
extern "C" void kernel_launch(void* const* d_in, const int* in_sizes, int n_in,
                              void* d_out, int out_size, void* d_ws, size_t ws_size,
                              hipStream_t stream) {
    (void)in_sizes; (void)n_in; (void)out_size; (void)ws_size;
    const float* X     = (const float*)d_in[0];
    const float* gamma = (const float*)d_in[1];
    const float* beta  = (const float*)d_in[2];
    const float* w_in  = (const float*)d_in[3];
    const float* w_out = (const float*)d_in[4];
    const float* b_out = (const float*)d_in[5];
    float* out = (float*)d_out;

    char* ws = (char*)d_ws;
    size_t off = 0;
    auto wsalloc = [&](size_t bytes) {
        size_t o = off;
        off += (bytes + 255) & ~(size_t)255;
        return o;
    };
    float*  stats  = (float*) (ws + wsalloc(32 * sizeof(float)));
    float*  qnorm  = (float*) (ws + wsalloc((size_t)NB * NN * sizeof(float)));
    float*  knorm  = (float*) (ws + wsalloc((size_t)NB * NN * sizeof(float)));
    float*  Avec   = (float*) (ws + wsalloc((size_t)NB * NC * sizeof(float)));
    bf16_t* win_bf = (bf16_t*)(ws + wsalloc((size_t)OC3 * NC * sizeof(bf16_t)));
    bf16_t* w2     = (bf16_t*)(ws + wsalloc((size_t)NB * NC * NC * sizeof(bf16_t)));
    bf16_t* xn     = (bf16_t*)(ws + wsalloc((size_t)NB * NN * NC * sizeof(bf16_t)));
    bf16_t* qkv    = (bf16_t*)(ws + wsalloc((size_t)NB * OC3 * NN * sizeof(bf16_t)));

    // 1) LayerNorm stats
    k_stats<<<dim3(NB), dim3(256), 0, stream>>>(X, stats);
    // 2) normalize + transpose to (b, n, c) bf16
    k_norm_t<<<dim3(NN / 32, NC / 32, NB), dim3(256), 0, stream>>>(X, gamma, beta, stats, xn);
    // 3) cast w_in
    k_cast_win<<<dim3((OC3 * NC + 255) / 256), dim3(256), 0, stream>>>(w_in, win_bf, OC3 * NC);
    // 4) GEMM1: qkv = w_in @ Xn
    k_gemm1<<<dim3(NN / 128, OC3 / 128, NB), dim3(256), 0, stream>>>(win_bf, xn, qkv);
    // 5) channel norms of Q, K
    k_qknorm<<<dim3(NN / 256, NB), dim3(256), 0, stream>>>(qkv, qnorm, knorm);
    // 6) A[b,c] = sum_n Khat*V
    k_avec<<<dim3(NC, NB), dim3(256), 0, stream>>>(qkv, knorm, Avec);
    // 7) B2 = Qhat transposed (reuse xn buffer)
    k_b2t<<<dim3(NN / 32, NC / 32, NB), dim3(256), 0, stream>>>(qkv, qnorm, xn);
    // 8) w2 = w_out * A
    k_w2<<<dim3((NB * NC * NC + 255) / 256), dim3(256), 0, stream>>>(w_out, Avec, w2);
    // 9) GEMM2 + bias + residual -> out
    k_gemm2<<<dim3(NN / 128, NC / 128, NB), dim3(256), 0, stream>>>(w2, xn, b_out, X, out);
}
